// Block_1829656068393
// MI455X (gfx1250) — matrix operand order repaired and checked by reading, not compiled, not run on hardware
//
#include <hip/hip_runtime.h>

// ---------------------------------------------------------------------------
// Transformer block for MI455X (gfx1250, wave32, WMMA f32_16x16x32_f16)
//   B=4, L=2048, D=1024, H=16, DH=64, DFF=4096
// ---------------------------------------------------------------------------

typedef __attribute__((ext_vector_type(16))) _Float16 v16h;
typedef __attribute__((ext_vector_type(8)))  _Float16 v8h;
typedef __attribute__((ext_vector_type(8)))  float    v8f;

constexpr int Bb   = 4;
constexpr int Ls   = 2048;
constexpr int Dd   = 1024;
constexpr int Hh   = 16;
constexpr int DHh  = 64;
constexpr int DFFf = 4096;
constexpr int Mrows = Bb * Ls;            // 8192 token rows

__device__ __forceinline__ v8f wmma32(v16h a, v16h b, v8f c) {
  // D = A(16x32 f16) * B(32x16 f16) + C(16x16 f32)
  return __builtin_amdgcn_wmma_f32_16x16x32_f16(false, a, false, b, (short)0, c,
                                                false, false);
}

// Load a 16-half fragment as two contiguous 8-half (16B) chunks, `gap` halves apart.
// A- and B-fragments both use gap=8 (contiguous 32B, k = 16*hi + s).
__device__ __forceinline__ v16h ld_pair16(const _Float16* p, int gap) {
  v16h r;
  *(v8h*)&r       = *(const v8h*)p;
  *((v8h*)&r + 1) = *(const v8h*)(p + gap);
  return r;
}

// ---------------------------------------------------------------------------
// Weight transpose + f32->f16 convert:  WT[n*K + k] = (f16) W[k*N + n]
// ---------------------------------------------------------------------------
__global__ void __launch_bounds__(256)
transpose_to_f16(const float* __restrict__ W, _Float16* __restrict__ WT,
                 int K, int N) {
  __shared__ float tile[32][33];
  const int kb = blockIdx.y * 32, nb = blockIdx.x * 32;
  const int tx = threadIdx.x, ty = threadIdx.y;  // block (32,8)
#pragma unroll
  for (int i = 0; i < 4; ++i)
    tile[ty + 8 * i][tx] = W[(long)(kb + ty + 8 * i) * N + nb + tx];
  __syncthreads();
#pragma unroll
  for (int i = 0; i < 4; ++i)
    WT[(long)(nb + ty + 8 * i) * K + kb + tx] = (_Float16)tile[tx][ty + 8 * i];
}

// ---------------------------------------------------------------------------
// LayerNorm over D=1024, one block per row, writes f16 activation matrix
// ---------------------------------------------------------------------------
__global__ void __launch_bounds__(256)
ln_to_f16(const float* __restrict__ X, const float* __restrict__ g,
          const float* __restrict__ be, _Float16* __restrict__ Out) {
  __shared__ float s1[256], s2[256];
  const int tid = threadIdx.x;
  const long row = blockIdx.x;
  const float* x = X + row * Dd;
  float v[4], s = 0.f, sq = 0.f;
#pragma unroll
  for (int i = 0; i < 4; ++i) {
    v[i] = x[tid + 256 * i];
    s += v[i];
    sq += v[i] * v[i];
  }
  s1[tid] = s; s2[tid] = sq;
  __syncthreads();
  for (int o = 128; o > 0; o >>= 1) {
    if (tid < o) { s1[tid] += s1[tid + o]; s2[tid] += s2[tid + o]; }
    __syncthreads();
  }
  const float mu  = s1[0] * (1.0f / Dd);
  const float var = s2[0] * (1.0f / Dd) - mu * mu;
  const float rs  = rsqrtf(var + 1e-5f);
#pragma unroll
  for (int i = 0; i < 4; ++i) {
    const int c = tid + 256 * i;
    Out[row * Dd + c] = (_Float16)((v[i] - mu) * rs * g[c] + be[c]);
  }
}

// ---------------------------------------------------------------------------
// WMMA GEMM:  C[M x N] = A[M x K] (f16) * BT[N x K]^T (f16) + bias, epilogue by MODE
//   MODE 0: write f16 to [B,H,L,DH]   (Q, K projections)
//   MODE 1: write f16 to [B,H,DH,L]   (V projection, pre-transposed for attention)
//   MODE 2: write f32  Out = Res + acc + bias   (O-proj residual / MLP2 residual)
//   MODE 3: write f16  relu(acc + bias)         (MLP1 hidden)
// Block: 256 thr = 8 waves; wave tile 32x64; block tile 128x128.
// ---------------------------------------------------------------------------
template <int MODE>
__global__ void __launch_bounds__(256)
gemm_wmma(const _Float16* __restrict__ A, const _Float16* __restrict__ BT,
          const float* __restrict__ bias, const float* __restrict__ Res,
          void* __restrict__ OutP, int Kd, int Nd) {
  const int tid  = threadIdx.x;
  const int lane = tid & 31, wave = tid >> 5;
  const int hi = lane >> 4, lo = lane & 15;
  const int wy = wave & 3, wx = wave >> 2;           // 4x2 wave grid
  const long rowA0 = (long)blockIdx.y * 128 + wy * 32;
  const long rowB0 = (long)blockIdx.x * 128 + wx * 64;

  const _Float16* Ar[2];
#pragma unroll
  for (int m = 0; m < 2; ++m) Ar[m] = A + (rowA0 + m * 16 + lo) * (long)Kd;
  const _Float16* Br[4];
#pragma unroll
  for (int j = 0; j < 4; ++j) Br[j] = BT + (rowB0 + j * 16 + lo) * (long)Kd;

  v8f acc[2][4] = {};

  for (int k = 0; k < Kd; k += 32) {
    if (k + 32 < Kd) {            // feed the next step into the cache hierarchy
      __builtin_prefetch(Ar[0] + k + 32 + hi * 8, 0, 0);
      __builtin_prefetch(Br[0] + k + 32 + hi * 16, 0, 0);
    }
    v16h af[2], bf[4];
#pragma unroll
    for (int m = 0; m < 2; ++m) af[m] = ld_pair16(Ar[m] + k + hi * 16, 8);
#pragma unroll
    for (int j = 0; j < 4; ++j) bf[j] = ld_pair16(Br[j] + k + hi * 16, 8);
#pragma unroll
    for (int m = 0; m < 2; ++m)
#pragma unroll
      for (int j = 0; j < 4; ++j) acc[m][j] = wmma32(af[m], bf[j], acc[m][j]);
  }

  float bj[4];
#pragma unroll
  for (int j = 0; j < 4; ++j) bj[j] = bias[rowB0 + j * 16 + lo];

#pragma unroll
  for (int m = 0; m < 2; ++m)
#pragma unroll
    for (int j = 0; j < 4; ++j)
#pragma unroll
      for (int r = 0; r < 8; ++r) {
        const long row = rowA0 + m * 16 + r + 8 * hi;   // token row in [0,8192)
        const long col = rowB0 + j * 16 + lo;
        const float v = acc[m][j][r] + bj[j];
        if constexpr (MODE == 0) {                      // [B,H,L,DH] f16
          _Float16* Out = (_Float16*)OutP;
          const long b = row >> 11, l = row & (Ls - 1);
          const long h = col >> 6, dh = col & 63;
          Out[((b * Hh + h) * Ls + l) * DHh + dh] = (_Float16)v;
        } else if constexpr (MODE == 1) {               // [B,H,DH,L] f16
          _Float16* Out = (_Float16*)OutP;
          const long b = row >> 11, l = row & (Ls - 1);
          const long h = col >> 6, dh = col & 63;
          Out[((b * Hh + h) * DHh + dh) * Ls + l] = (_Float16)v;
        } else if constexpr (MODE == 2) {               // residual f32
          float* Out = (float*)OutP;
          const long idx = row * Nd + col;
          Out[idx] = Res[idx] + v;
        } else {                                        // relu f16
          _Float16* Out = (_Float16*)OutP;
          Out[row * (long)Nd + col] = (_Float16)(v > 0.f ? v : 0.f);
        }
      }
}

// ---------------------------------------------------------------------------
// Causal flash attention. One wave per 16-query tile; 8 waves / block.
// Q,K: [B,H,L,DH] f16; VT: [B,H,DH,L] f16; AV out: [B,L,H*DH] f16.
// Scores: 4 WMMAs per 32 keys (K-dim = DH = 64); P routed through LDS to
// become an A-layout f16 fragment; P*V: 4 WMMAs per 32 keys.
// ---------------------------------------------------------------------------
__global__ void __launch_bounds__(256)
attn_kernel(const _Float16* __restrict__ Q, const _Float16* __restrict__ Kx,
            const _Float16* __restrict__ VT, _Float16* __restrict__ AV) {
  __shared__ _Float16 Plds[8][16 * 32];                 // 1KB per wave
  const int tid = threadIdx.x, lane = tid & 31, wave = tid >> 5;
  const int hi = lane >> 4, lo = lane & 15;
  const int bh = blockIdx.y;
  const int q0 = blockIdx.x * 128 + wave * 16;          // wave-uniform

  const _Float16* Qb = Q  + (long)bh * Ls * DHh;
  const _Float16* Kb = Kx + (long)bh * Ls * DHh;
  const _Float16* Vb = VT + (long)bh * DHh * Ls;

  const _Float16* qrow = Qb + (long)(q0 + lo) * DHh;
  const v16h aQ0 = ld_pair16(qrow + hi * 16, 8);        // dh 0..31
  const v16h aQ1 = ld_pair16(qrow + 32 + hi * 16, 8);   // dh 32..63

  v8f oacc[4] = {};
  float mrow[8], lrow[8];
#pragma unroll
  for (int r = 0; r < 8; ++r) { mrow[r] = -1e30f; lrow[r] = 0.f; }

  const float scale = 0.125f;                           // 1/sqrt(64)
  const int kend = (q0 + 16 + 31) & ~31;                // keys 0 .. q0+15, step 32

  for (int kb = 0; kb < kend; kb += 32) {
    // ---- scores: two 16x16 tiles covering keys [kb,kb+16) and [kb+16,kb+32)
    v8f s0 = {}, s1 = {};
    const _Float16* kr0 = Kb + (long)(kb + lo) * DHh + hi * 16;
    const _Float16* kr1 = Kb + (long)(kb + 16 + lo) * DHh + hi * 16;
    s0 = wmma32(aQ0, ld_pair16(kr0, 8), s0);
    s0 = wmma32(aQ1, ld_pair16(kr0 + 32, 8), s0);
    s1 = wmma32(aQ0, ld_pair16(kr1, 8), s1);
    s1 = wmma32(aQ1, ld_pair16(kr1 + 32, 8), s1);

    // ---- causal mask + online softmax (rows live in 16-lane halves)
    float p0[8], p1[8], alpha[8];
#pragma unroll
    for (int r = 0; r < 8; ++r) {
      const int qr = q0 + r + 8 * hi;
      const int c0 = kb + lo, c1 = c0 + 16;
      const float v0 = (c0 <= qr) ? s0[r] * scale : -1e30f;
      const float v1 = (c1 <= qr) ? s1[r] * scale : -1e30f;
      float t = fmaxf(v0, v1);
      t = fmaxf(t, __shfl_xor(t, 1, 32));
      t = fmaxf(t, __shfl_xor(t, 2, 32));
      t = fmaxf(t, __shfl_xor(t, 4, 32));
      t = fmaxf(t, __shfl_xor(t, 8, 32));
      const float mn = fmaxf(mrow[r], t);
      const float a  = __expf(mrow[r] - mn);            // 0 on first touch
      const float e0 = __expf(v0 - mn);                 // 0 when masked
      const float e1 = __expf(v1 - mn);
      float ps = e0 + e1;
      ps += __shfl_xor(ps, 1, 32);
      ps += __shfl_xor(ps, 2, 32);
      ps += __shfl_xor(ps, 4, 32);
      ps += __shfl_xor(ps, 8, 32);
      lrow[r] = lrow[r] * a + ps;
      mrow[r] = mn;
      alpha[r] = a;
      p0[r] = e0;
      p1[r] = e1;
    }
#pragma unroll
    for (int j = 0; j < 4; ++j)
#pragma unroll
      for (int r = 0; r < 8; ++r) oacc[j][r] *= alpha[r];

    // ---- C-layout f32 P -> LDS -> A-layout f16 fragment (wave-private)
#pragma unroll
    for (int r = 0; r < 8; ++r) {
      Plds[wave][(r + 8 * hi) * 32 + lo]      = (_Float16)p0[r];
      Plds[wave][(r + 8 * hi) * 32 + 16 + lo] = (_Float16)p1[r];
    }
    asm volatile("s_wait_dscnt 0" ::: "memory");
    v16h aP;
    {
      const int base = lo * 32 + hi * 16;
#pragma unroll
      for (int e = 0; e < 8; ++e) aP[e] = Plds[wave][base + e];
#pragma unroll
      for (int e = 0; e < 8; ++e) aP[8 + e] = Plds[wave][base + 8 + e];
    }

    // ---- O += P(16x32) * V(32x64): VT rows are dh, contiguous along keys
#pragma unroll
    for (int j = 0; j < 4; ++j) {
      const _Float16* vr = Vb + (long)(j * 16 + lo) * Ls + kb + hi * 16;
      oacc[j] = wmma32(aP, ld_pair16(vr, 8), oacc[j]);
    }
  }

  // ---- normalize and scatter to [B, L, H*DH] f16
  const int b = bh >> 4, h = bh & 15;
#pragma unroll
  for (int j = 0; j < 4; ++j)
#pragma unroll
    for (int r = 0; r < 8; ++r) {
      const int qr = q0 + r + 8 * hi;
      const int dh = j * 16 + lo;
      const float val = oacc[j][r] / lrow[r];
      AV[((long)(b * Ls + qr)) * Dd + h * DHh + dh] = (_Float16)val;
    }
}

// ---------------------------------------------------------------------------
// Host orchestration (all on `stream`; scratch ~226 MB in d_ws)
// ---------------------------------------------------------------------------
extern "C" void kernel_launch(void* const* d_in, const int* in_sizes, int n_in,
                              void* d_out, int out_size, void* d_ws,
                              size_t ws_size, hipStream_t stream) {
  const float* X   = (const float*)d_in[0];
  const float* WQ  = (const float*)d_in[1];
  const float* bQ  = (const float*)d_in[2];
  const float* WK  = (const float*)d_in[3];
  const float* bK  = (const float*)d_in[4];
  const float* WV  = (const float*)d_in[5];
  const float* bV  = (const float*)d_in[6];
  const float* WO  = (const float*)d_in[7];
  const float* bO  = (const float*)d_in[8];
  const float* W1  = (const float*)d_in[9];
  const float* b1  = (const float*)d_in[10];
  const float* W2  = (const float*)d_in[11];
  const float* b2  = (const float*)d_in[12];
  const float* g1  = (const float*)d_in[13];
  const float* be1 = (const float*)d_in[14];
  const float* g2  = (const float*)d_in[15];
  const float* be2 = (const float*)d_in[16];
  float* Out = (float*)d_out;

  char* ws = (char*)d_ws;
  size_t off = 0;
  auto alloc = [&](size_t bytes) -> void* {
    void* p = ws + off;
    off += (bytes + 255) & ~(size_t)255;
    return p;
  };

  _Float16* WQt = (_Float16*)alloc((size_t)Dd * Dd * 2);
  _Float16* WKt = (_Float16*)alloc((size_t)Dd * Dd * 2);
  _Float16* WVt = (_Float16*)alloc((size_t)Dd * Dd * 2);
  _Float16* WOt = (_Float16*)alloc((size_t)Dd * Dd * 2);
  _Float16* W1t = (_Float16*)alloc((size_t)Dd * DFFf * 2);
  _Float16* W2t = (_Float16*)alloc((size_t)DFFf * Dd * 2);
  _Float16* Xn16  = (_Float16*)alloc((size_t)Mrows * Dd * 2);
  _Float16* Qb    = (_Float16*)alloc((size_t)Mrows * Dd * 2);
  _Float16* Kb    = (_Float16*)alloc((size_t)Mrows * Dd * 2);
  _Float16* VTb   = (_Float16*)alloc((size_t)Mrows * Dd * 2);
  _Float16* AV16  = (_Float16*)alloc((size_t)Mrows * Dd * 2);
  float*    Yb    = (float*)alloc((size_t)Mrows * Dd * 4);
  _Float16* Xn2   = (_Float16*)alloc((size_t)Mrows * Dd * 2);
  _Float16* H16   = (_Float16*)alloc((size_t)Mrows * DFFf * 2);
  (void)ws_size; (void)n_in; (void)in_sizes; (void)out_size;

  const dim3 tb(32, 8);
  transpose_to_f16<<<dim3(Dd / 32, Dd / 32), tb, 0, stream>>>(WQ, WQt, Dd, Dd);
  transpose_to_f16<<<dim3(Dd / 32, Dd / 32), tb, 0, stream>>>(WK, WKt, Dd, Dd);
  transpose_to_f16<<<dim3(Dd / 32, Dd / 32), tb, 0, stream>>>(WV, WVt, Dd, Dd);
  transpose_to_f16<<<dim3(Dd / 32, Dd / 32), tb, 0, stream>>>(WO, WOt, Dd, Dd);
  transpose_to_f16<<<dim3(DFFf / 32, Dd / 32), tb, 0, stream>>>(W1, W1t, Dd, DFFf);
  transpose_to_f16<<<dim3(Dd / 32, DFFf / 32), tb, 0, stream>>>(W2, W2t, DFFf, Dd);

  ln_to_f16<<<Mrows, 256, 0, stream>>>(X, g1, be1, Xn16);

  const dim3 gD(Dd / 128, Mrows / 128);                 // (8, 64)
  gemm_wmma<0><<<gD, 256, 0, stream>>>(Xn16, WQt, bQ, nullptr, Qb, Dd, Dd);
  gemm_wmma<0><<<gD, 256, 0, stream>>>(Xn16, WKt, bK, nullptr, Kb, Dd, Dd);
  gemm_wmma<1><<<gD, 256, 0, stream>>>(Xn16, WVt, bV, nullptr, VTb, Dd, Dd);

  attn_kernel<<<dim3(Ls / 128, Bb * Hh), 256, 0, stream>>>(Qb, Kb, VTb, AV16);

  gemm_wmma<2><<<gD, 256, 0, stream>>>(AV16, WOt, bO, X, Yb, Dd, Dd);
  ln_to_f16<<<Mrows, 256, 0, stream>>>(Yb, g2, be2, Xn2);
  gemm_wmma<3><<<dim3(DFFf / 128, Mrows / 128), 256, 0, stream>>>(
      Xn2, W1t, b1, nullptr, H16, Dd, DFFf);
  gemm_wmma<2><<<gD, 256, 0, stream>>>(H16, W2t, b2, Yb, Out, DFFf, Dd);
}
